// MultiHeadAttention_2422361555349
// MI455X (gfx1250) — matrix operand order repaired and checked by reading, not compiled, not run on hardware
//
#include <hip/hip_runtime.h>

// ---------------------------------------------------------------------------
// MultiHeadAttention forward for MI455X (gfx1250, wave32, WMMA).
//   q = query@Wq^T+bq ; k = key@Wk^T+bk ; v = value@Wv^T+bv   (f16 out)
//   flash attention per (b,h), 64-query tiles, f32 online softmax,
//   K/V tiles staged via GLOBAL_LOAD_ASYNC_TO_LDS (ASYNCcnt, double-buffered)
//   out = ctx@Wo^T+bo (f32 out)
// Workspace layout (needs 64 MB): qf16 | kf16 | vT_f16 | ctx_f16, 16 MB each.
// All matrix math via v_wmma_f32_16x16x32_f16 (f32 accumulate).
// ---------------------------------------------------------------------------

#define B_   4
#define S_   2048
#define D_   1024
#define H_   16
#define DK_  64
#define M_   (B_ * S_)   // 8192 rows for the projection GEMMs

typedef __attribute__((ext_vector_type(16))) _Float16 v16h;
typedef __attribute__((ext_vector_type(8)))  float    v8f;

union AFrag {            // 16 halfs per lane = one 16x32 f16 WMMA operand
    v16h v;
    uint4 q[2];
};

static __device__ inline v8f wmma_f16(const v16h a, const v16h b, const v8f c) {
    // 8 args: (neg_a, A, neg_b, B, c_mod, C, reuse_a, reuse_b)
    return __builtin_amdgcn_wmma_f32_16x16x32_f16(false, a, false, b,
                                                  (short)0, c, false, false);
}

// Async copy: 16 bytes per lane, global (saddr + 32-bit voffset) -> LDS.
// vdst = per-lane LDS byte address; tracked by ASYNCcnt (manual waits).
static __device__ inline void async_cp_b128(unsigned lds_addr, unsigned goff,
                                            const void* sbase) {
    asm volatile("global_load_async_to_lds_b128 %0, %1, %2"
                 :: "v"(lds_addr), "v"(goff), "s"(sbase) : "memory");
}
static __device__ inline void wait_async(void) {
    asm volatile("s_wait_asynccnt 0" ::: "memory");
}
static __device__ inline void wait_async_le4(void) {
    asm volatile("s_wait_asynccnt 4" ::: "memory");
}
static __device__ inline unsigned lds_off(const void* p) {
    // generic pointer to LDS: low 32 bits are the LDS byte address
    return (unsigned)(unsigned long long)p;
}

// ---------------------------------------------------------------------------
// Projection GEMM: C[M x 1024] = X[M x 1024] @ W^T + bias
// Block: 128 threads (4 waves). Block tile: 64(M) x 64(N). K step: 32.
// mode 0: store f16 row-major [M, D]          (Q, K projections)
// mode 1: store f16 transposed [B, H, DK, S]  (V projection -> V^T)
// mode 2: store f32 row-major [M, D]          (final output projection)
// ---------------------------------------------------------------------------
template <typename XT>
__global__ __launch_bounds__(128) void proj_kernel(
    const XT* __restrict__ X, const float* __restrict__ W,
    const float* __restrict__ bias, void* __restrict__ out, int mode)
{
    __shared__ _Float16 Xs[64 * 32];   // [m][k] halfs
    __shared__ _Float16 Ws[64 * 32];   // [n][k] halfs (== B operand N-major)

    const int tid   = threadIdx.x;
    const int wave  = tid >> 5;
    const int lane  = tid & 31;
    const int lh    = lane & 15;   // 0..15
    const int lg    = lane >> 4;   // 0 or 1
    const int mbase = blockIdx.x * 64;
    const int nbase = blockIdx.y * 64;
    const unsigned xs_base = lds_off(&Xs[0]);

    v8f acc[4] = {};

    for (int k0 = 0; k0 < D_; k0 += 32) {
        // ---- stage X tile ----
        if constexpr (sizeof(XT) == 4) {       // f32 input: load+convert
            for (int rep = 0; rep < 4; ++rep) {
                int id = tid + rep * 128;      // 0..511
                int r  = id >> 3;
                int c  = (id & 7) * 4;
                float4 xf = *(const float4*)(X + (size_t)(mbase + r) * D_ + k0 + c);
                _Float16* xp = &Xs[r * 32 + c];
                xp[0] = (_Float16)xf.x; xp[1] = (_Float16)xf.y;
                xp[2] = (_Float16)xf.z; xp[3] = (_Float16)xf.w;
            }
        } else {                               // f16 input: async copy to LDS
            for (int rep = 0; rep < 2; ++rep) {
                int id = tid + rep * 128;      // 0..255
                int r  = id >> 2;
                int c  = (id & 3) * 8;
                async_cp_b128(xs_base + (unsigned)(r * 32 + c) * 2,
                              (unsigned)(((mbase + r) * D_ + k0 + c) * 2), X);
            }
        }
        // ---- stage W tile (f32 -> f16): 64 rows x 32 cols ----
        for (int rep = 0; rep < 4; ++rep) {
            int id = tid + rep * 128;          // 0..511
            int r  = id >> 3;
            int c  = (id & 7) * 4;
            float4 wf = *(const float4*)(W + (size_t)(nbase + r) * D_ + k0 + c);
            _Float16* wp = &Ws[r * 32 + c];
            wp[0] = (_Float16)wf.x; wp[1] = (_Float16)wf.y;
            wp[2] = (_Float16)wf.z; wp[3] = (_Float16)wf.w;
        }
        if constexpr (sizeof(XT) == 2) wait_async();
        __syncthreads();

        // ---- A fragment for this wave's 16 rows ----
        AFrag a;
        const int arow = wave * 16 + lh;
        a.q[0] = *(const uint4*)&Xs[arow * 32 + lg * 8];
        a.q[1] = *(const uint4*)&Xs[arow * 32 + lg * 8 + 16];

        // ---- 4 N-tiles of 16 ----
        for (int t = 0; t < 4; ++t) {
            AFrag bf;
            const int nrow = t * 16 + lh;
            bf.q[0] = *(const uint4*)&Ws[nrow * 32 + lg * 8];
            bf.q[1] = *(const uint4*)&Ws[nrow * 32 + lg * 8 + 16];
            acc[t] = wmma_f16(a.v, bf.v, acc[t]);
        }
        __syncthreads();
    }

    // ---- epilogue: bias add + store (C layout: row = comp + 8*lg, col = lh)
    for (int t = 0; t < 4; ++t) {
        const int   gn = nbase + t * 16 + lh;
        const float bv = bias[gn];
        for (int c = 0; c < 8; ++c) {
            const int gm  = mbase + wave * 16 + c + lg * 8;
            const float v = acc[t][c] + bv;
            if (mode == 0) {
                ((_Float16*)out)[(size_t)gm * D_ + gn] = (_Float16)v;
            } else if (mode == 1) {
                const int b = gm >> 11;          // / S_
                const int s = gm & (S_ - 1);
                const int h = gn >> 6;           // / DK_
                const int d = gn & (DK_ - 1);
                ((_Float16*)out)[((size_t)(b * H_ + h) * DK_ + d) * S_ + s] =
                    (_Float16)v;
            } else {
                ((float*)out)[(size_t)gm * D_ + gn] = v;
            }
        }
    }
}

// ---------------------------------------------------------------------------
// Flash attention. Grid: (S/64, B*H). Block: 128 threads (4 waves).
// Each wave: 16 query rows; kv loop step 32; K/V tiles double-buffered in
// LDS via async-to-LDS copies (4 async ops per wave per tile -> wait <= 4).
//   Qf: [B, S, D] f16 (head slice contiguous)  -> A operand
//   Kf: [B, S, D] f16 (rows kv, contiguous dk) -> B operand for Q@K^T
//   Vt: [B, H, DK, S] f16                      -> B operand for P@V
//   ctx:[B, S, D] f16
// ---------------------------------------------------------------------------
__global__ __launch_bounds__(128) void attn_kernel(
    const _Float16* __restrict__ Qf, const _Float16* __restrict__ Kf,
    const _Float16* __restrict__ Vt, _Float16* __restrict__ ctx)
{
    __shared__ _Float16 Ks [2][32 * 64];   // [buf][kv][dk]
    __shared__ _Float16 Vts[2][64 * 32];   // [buf][dk][kv]
    __shared__ _Float16 Ps [4][16 * 32];   // per-wave P transpose buffer

    const int tid  = threadIdx.x;
    const int wave = tid >> 5;
    const int lane = tid & 31;
    const int lh   = lane & 15;
    const int lg   = lane >> 4;
    const int qbase = blockIdx.x * 64;
    const int bh    = blockIdx.y;
    const int b     = bh >> 4;
    const int h     = bh & 15;

    const _Float16* kptr = Kf + (size_t)b * S_ * D_ + h * DK_;
    const _Float16* vptr = Vt + (size_t)(b * H_ + h) * DK_ * S_;

    // per-thread staging chunk coordinates (8 halfs = 16B per chunk)
    int kr[2], kc[2], vr[2], vc[2];
    unsigned ksl[2][2], vsl[2][2];   // [rep][buf] LDS byte addresses
    for (int rep = 0; rep < 2; ++rep) {
        int id = tid + rep * 128;               // 0..255
        kr[rep] = id >> 3; kc[rep] = (id & 7) * 8;   // K: 8 chunks/row
        vr[rep] = id >> 2; vc[rep] = (id & 3) * 8;   // Vt: 4 chunks/row
        for (int bf = 0; bf < 2; ++bf) {
            ksl[rep][bf] = lds_off(&Ks [bf][kr[rep] * 64 + kc[rep]]);
            vsl[rep][bf] = lds_off(&Vts[bf][vr[rep] * 32 + vc[rep]]);
        }
    }
    auto stage = [&](int kv0, int bf) {
        for (int rep = 0; rep < 2; ++rep) {
            async_cp_b128(ksl[rep][bf],
                          (unsigned)(((kv0 + kr[rep]) * D_ + kc[rep]) * 2), kptr);
            async_cp_b128(vsl[rep][bf],
                          (unsigned)((vr[rep] * S_ + kv0 + vc[rep]) * 2), vptr);
        }
    };

    // ---- load Q A-fragments (dk=64 -> 2 k-steps of 32) ----
    AFrag qa[2];
    {
        const size_t qrow =
            ((size_t)(b * S_ + qbase + wave * 16 + lh)) * D_ + h * DK_;
        for (int j = 0; j < 2; ++j) {
            qa[j].q[0] = *(const uint4*)(Qf + qrow + j * 32 + lg * 8);
            qa[j].q[1] = *(const uint4*)(Qf + qrow + j * 32 + lg * 8 + 16);
        }
    }

    v8f o[4] = {};
    float rm[8], ls[8];
    for (int c = 0; c < 8; ++c) { rm[c] = -3.0e38f; ls[c] = 0.0f; }
    const float scale = 0.125f;   // 1/sqrt(64)

    stage(0, 0);                  // prologue: tile 0 in flight
    const int NT = S_ / 32;       // 64 kv tiles

    for (int it = 0; it < NT; ++it) {
        // issue next tile into the other buffer (clamped re-issue at the end:
        // harmless, writes a buffer nobody reads again)
        const int nxt = (it + 1 < NT) ? (it + 1) : it;
        stage(nxt * 32, (it + 1) & 1);
        wait_async_le4();         // tile `it` complete (in-order completion)
        __syncthreads();          // publish LDS tile to all waves

        const _Float16* Ksb = Ks [it & 1];
        const _Float16* Vtb = Vts[it & 1];

        // ---- scores: S[16 x 32] = Q(16x64) @ K^T ----
        v8f s0 = {}, s1 = {};
        for (int j = 0; j < 2; ++j) {
            AFrag b0, b1;
            b0.q[0] = *(const uint4*)&Ksb[(     lh) * 64 + j * 32 + lg * 8];
            b0.q[1] = *(const uint4*)&Ksb[(     lh) * 64 + j * 32 + lg * 8 + 16];
            b1.q[0] = *(const uint4*)&Ksb[(16 + lh) * 64 + j * 32 + lg * 8];
            b1.q[1] = *(const uint4*)&Ksb[(16 + lh) * 64 + j * 32 + lg * 8 + 16];
            s0 = wmma_f16(qa[j].v, b0.v, s0);
            s1 = wmma_f16(qa[j].v, b1.v, s1);
        }

        // ---- online softmax (row = comp + 8*lg, col = lane%16) ----
        float nm[8], f[8];
        for (int c = 0; c < 8; ++c) { s0[c] *= scale; s1[c] *= scale; }
        for (int c = 0; c < 8; ++c) {
            float m = fmaxf(s0[c], s1[c]);
            for (int off = 8; off >= 1; off >>= 1)
                m = fmaxf(m, __shfl_xor(m, off, 16));
            nm[c] = fmaxf(rm[c], m);
            f[c]  = __expf(rm[c] - nm[c]);
        }
        for (int c = 0; c < 8; ++c) {
            float p0 = __expf(s0[c] - nm[c]);
            float p1 = __expf(s1[c] - nm[c]);
            s0[c] = p0; s1[c] = p1;
            float ps = p0 + p1;
            for (int off = 8; off >= 1; off >>= 1)
                ps += __shfl_xor(ps, off, 16);
            ls[c] = ls[c] * f[c] + ps;
            rm[c] = nm[c];
        }
        for (int t = 0; t < 4; ++t)
            for (int c = 0; c < 8; ++c) o[t][c] *= f[c];

        // ---- P: C-layout f32 -> A-layout f16 via per-wave LDS buffer ----
        _Float16* Pw = Ps[wave];
        for (int c = 0; c < 8; ++c) {
            const int m = c + lg * 8;
            Pw[m * 32 + lh]      = (_Float16)s0[c];
            Pw[m * 32 + 16 + lh] = (_Float16)s1[c];
        }
        asm volatile("s_wait_dscnt 0" ::: "memory");   // LDS RAW safety
        AFrag pa;
        pa.q[0] = *(const uint4*)&Pw[lh * 32 + lg * 8];
        pa.q[1] = *(const uint4*)&Pw[lh * 32 + lg * 8 + 16];

        // ---- O += P @ V : 4 N-tiles over dk ----
        for (int t = 0; t < 4; ++t) {
            AFrag vf;
            vf.q[0] = *(const uint4*)&Vtb[(t * 16 + lh) * 32 + lg * 8];
            vf.q[1] = *(const uint4*)&Vtb[(t * 16 + lh) * 32 + lg * 8 + 16];
            o[t] = wmma_f16(pa.v, vf.v, o[t]);
        }
        __syncthreads();          // done reading this buffer
    }

    // ---- epilogue: normalize and store context (f16, [B,S,D]) ----
    for (int t = 0; t < 4; ++t) {
        for (int c = 0; c < 8; ++c) {
            const float val = o[t][c] / ls[c];
            const int   m   = qbase + wave * 16 + c + lg * 8;
            ctx[((size_t)(b * S_ + m)) * D_ + h * DK_ + t * 16 + lh] =
                (_Float16)val;
        }
    }
}

// ---------------------------------------------------------------------------
extern "C" void kernel_launch(void* const* d_in, const int* in_sizes, int n_in,
                              void* d_out, int out_size, void* d_ws, size_t ws_size,
                              hipStream_t stream)
{
    (void)in_sizes; (void)n_in; (void)out_size; (void)ws_size;

    const float* query = (const float*)d_in[0];
    const float* key   = (const float*)d_in[1];
    const float* value = (const float*)d_in[2];
    const float* Wq    = (const float*)d_in[3];
    const float* bq    = (const float*)d_in[4];
    const float* Wk    = (const float*)d_in[5];
    const float* bk    = (const float*)d_in[6];
    const float* Wv    = (const float*)d_in[7];
    const float* bv    = (const float*)d_in[8];
    const float* Wo    = (const float*)d_in[9];
    const float* bo    = (const float*)d_in[10];

    const size_t N = (size_t)M_ * D_;           // 8388608 elements
    _Float16* qf  = (_Float16*)d_ws;
    _Float16* kf  = qf  + N;
    _Float16* vt  = kf  + N;
    _Float16* ctx = vt  + N;

    dim3 pgrid(M_ / 64, D_ / 64);   // 128 x 16
    dim3 pblk(128);
    proj_kernel<float><<<pgrid, pblk, 0, stream>>>(query, Wq, bq, (void*)qf, 0);
    proj_kernel<float><<<pgrid, pblk, 0, stream>>>(key,   Wk, bk, (void*)kf, 0);
    proj_kernel<float><<<pgrid, pblk, 0, stream>>>(value, Wv, bv, (void*)vt, 1);

    dim3 agrid(S_ / 64, B_ * H_);   // 32 x 64
    attn_kernel<<<agrid, pblk, 0, stream>>>(qf, kf, vt, ctx);

    proj_kernel<_Float16><<<pgrid, pblk, 0, stream>>>(ctx, Wo, bo, d_out, 2);
}
